// SimpleEncoder_9749575762050
// MI455X (gfx1250) — compile-verified, matrix operand order checked by reading, not run on hardware
//
#include <hip/hip_runtime.h>
#include <hip/hip_bf16.h>
#include <math.h>

typedef __attribute__((ext_vector_type(16))) _Float16 v16h;
typedef __attribute__((ext_vector_type(8)))  _Float16 v8h;
typedef __attribute__((ext_vector_type(8)))  float    v8f;

#define LN_EPS 1e-5f

__device__ __forceinline__ float silu_f(float x)  { return x / (1.f + __expf(-x)); }
__device__ __forceinline__ float softplus_f(float x) { return (x > 20.f) ? x : log1pf(__expf(x)); }

// ---------------------------------------------------------------------------
// WMMA fragment load: 16x32 f16 fragment from an f16 row-major matrix,
// row stride K (elements). CDNA5 layout (ISA 7.12.2, 16-bit A 16x32):
//   lanes 0-15 : row=lane,    elems 0..7 = K[k0..k0+7],   elems 8..15 = K[k0+16..k0+23]
//   lanes 16-31: row=lane-16, elems 0..7 = K[k0+8..+15],  elems 8..15 = K[k0+24..+31]
// Two 16-byte loads per lane. Invalid rows are clamped to row 0 (memory-safe);
// resulting garbage only reaches out-of-range outputs, which are never stored.
// ---------------------------------------------------------------------------
__device__ __forceinline__ v16h load_frag_h(const _Float16* __restrict__ base,
                                            int row, int K, int k0, int half)
{
    const _Float16* __restrict__ p = base + (size_t)row * (size_t)K;
    const v8h lo = *(const v8h*)(p + k0 + 8 * half);
    const v8h hi = *(const v8h*)(p + k0 + 16 + 8 * half);
    v16h f;
#pragma unroll
    for (int e = 0; e < 8; ++e) { f[e] = lo[e]; f[8 + e] = hi[e]; }
    return f;
}

// ---------------------------------------------------------------------------
// GEMM: D[M,N] = A[M,K] * W[N,K]^T  (f16 in, fp32 WMMA acc)
// One wave computes a 16x64 strip: A fragment reused across 4 N-tiles,
// 4 v_wmma issues per 32-wide K step. K multiple of 32 (true for all uses).
// epi==0: Out[m*N+n] = acc
// epi==1: residual scatter-add into NCHW Xres[B][resC][L], token m = b*L + l
// ---------------------------------------------------------------------------
__global__ __launch_bounds__(128)
void wmma_gemm_kernel(const _Float16* __restrict__ A, const _Float16* __restrict__ W,
                      float* __restrict__ Out, float* __restrict__ Xres,
                      int M, int N, int K, int L, int resC, int epi)
{
    const int tiles_n  = (N + 15) >> 4;
    const int strips_n = (tiles_n + 3) >> 2;          // 4 N-tiles per wave
    const int tiles_m  = (M + 15) >> 4;
    const int nstrip   = tiles_m * strips_n;
    const int strip    = blockIdx.x * (blockDim.x >> 5) + (threadIdx.x >> 5);
    if (strip >= nstrip) return;                      // wave-uniform: EXEC stays full

    const int lane = threadIdx.x & 31;
    const int tm = strip / strips_n;
    const int sn = strip - tm * strips_n;
    const int half = lane >> 4;
    const int lr   = lane & 15;

    const int m  = tm * 16 + lr;                      // A-fragment row for this lane
    const int ma = (m < M) ? m : 0;

    int nrow[4];
#pragma unroll
    for (int t = 0; t < 4; ++t) {
        const int n = (sn * 4 + t) * 16 + lr;         // B-fragment column for this lane
        nrow[t] = (n < N) ? n : 0;
    }

    v8f acc0 = {}, acc1 = {}, acc2 = {}, acc3 = {};
    for (int k0 = 0; k0 < K; k0 += 32) {
        const v16h af = load_frag_h(A, ma, K, k0, half);
        const v16h b0 = load_frag_h(W, nrow[0], K, k0, half);
        const v16h b1 = load_frag_h(W, nrow[1], K, k0, half);
        const v16h b2 = load_frag_h(W, nrow[2], K, k0, half);
        const v16h b3 = load_frag_h(W, nrow[3], K, k0, half);
        acc0 = __builtin_amdgcn_wmma_f32_16x16x32_f16(false, af, false, b0, (short)0, acc0, false, false);
        acc1 = __builtin_amdgcn_wmma_f32_16x16x32_f16(false, af, false, b1, (short)0, acc1, false, false);
        acc2 = __builtin_amdgcn_wmma_f32_16x16x32_f16(false, af, false, b2, (short)0, acc2, false, false);
        acc3 = __builtin_amdgcn_wmma_f32_16x16x32_f16(false, af, false, b3, (short)0, acc3, false, false);
    }

    // C/D layout: VGPR r -> row tm*16 + r + 8*half, col = tile_n*16 + (lane&15)
    const int row0 = tm * 16 + 8 * half;
#pragma unroll
    for (int t = 0; t < 4; ++t) {
        const v8f acc = (t == 0) ? acc0 : (t == 1) ? acc1 : (t == 2) ? acc2 : acc3;
        const int col = (sn * 4 + t) * 16 + lr;
        if (col < N) {
#pragma unroll
            for (int r = 0; r < 8; ++r) {
                const int mm = row0 + r;
                if (mm < M) {
                    const float v = acc[r];
                    if (epi == 0) {
                        Out[(size_t)mm * N + col] = v;
                    } else {
                        const int b = mm / L;
                        const int l = mm - b * L;
                        Xres[((size_t)b * resC + col) * (size_t)L + l] += v;
                    }
                }
            }
        }
    }
}

static inline void launch_gemm(const _Float16* A, const _Float16* W, float* Out, float* Xres,
                               int M, int N, int K, int L, int resC, int epi,
                               hipStream_t s)
{
    const int tiles_n  = (N + 15) / 16;
    const int strips_n = (tiles_n + 3) / 4;
    const int nstrip   = ((M + 15) / 16) * strips_n;
    const int blocks   = (nstrip + 3) / 4;            // 4 waves / 128-thread block
    wmma_gemm_kernel<<<blocks, 128, 0, s>>>(A, W, Out, Xres, M, N, K, L, resC, epi);
}

// ---------------------------------------------------------------------------
// fp32 -> f16 weight conversion (once per block per launch; weights are tiny)
// ---------------------------------------------------------------------------
__global__ __launch_bounds__(256)
void f32_to_f16_kernel(const float* __restrict__ src, _Float16* __restrict__ dst, int n)
{
    const int i = blockIdx.x * blockDim.x + threadIdx.x;
    if (i < n) dst[i] = (_Float16)src[i];
}

// ---------------------------------------------------------------------------
// LayerNorm over channels of NCHW tensor -> token-major f16 [B*L, C]
// one wave32 per token, shuffle reduction
// ---------------------------------------------------------------------------
__global__ __launch_bounds__(256)
void ln_kernel(const float* __restrict__ x, const float* __restrict__ g,
               const float* __restrict__ b, _Float16* __restrict__ s,
               int tokens, int C, int L)
{
    const int wave = (blockIdx.x * blockDim.x + threadIdx.x) >> 5;
    const int lane = threadIdx.x & 31;
    if (wave >= tokens) return;
    const int bb = wave / L;
    const int l  = wave - bb * L;
    const float* __restrict__ xp = x + (size_t)bb * C * L + l;   // stride L per channel

    float sum = 0.f, sum2 = 0.f;
    for (int c = lane; c < C; c += 32) {
        const float v = xp[(size_t)c * L];
        sum += v; sum2 += v * v;
    }
#pragma unroll
    for (int off = 16; off > 0; off >>= 1) {
        sum  += __shfl_down(sum,  off);
        sum2 += __shfl_down(sum2, off);
    }
    sum  = __shfl(sum,  0);
    sum2 = __shfl(sum2, 0);
    const float mean = sum / (float)C;
    const float var  = sum2 / (float)C - mean * mean;
    const float inv  = rsqrtf(var + LN_EPS);

    _Float16* __restrict__ sp = s + (size_t)wave * C;
    for (int c = lane; c < C; c += 32) {
        const float v = xp[(size_t)c * L];
        sp[c] = (_Float16)((v - mean) * inv * g[c] + b[c]);
    }
}

// ---------------------------------------------------------------------------
// Causal depthwise conv1d (kernel 4, left pad 3) + bias + SiLU.
// Writes fp32 (scan input) and f16 (GEMM2 operand) views of xh.
// ---------------------------------------------------------------------------
__global__ __launch_bounds__(256)
void dwconv_silu_kernel(const float* __restrict__ xz, const float* __restrict__ cw,
                        const float* __restrict__ cb, float* __restrict__ xh,
                        _Float16* __restrict__ xhh, int B, int L, int di)
{
    const int idx = blockIdx.x * blockDim.x + threadIdx.x;
    const int total = B * L * di;
    if (idx >= total) return;
    const int d = idx % di;
    const int t = idx / di;            // token = b*L + l
    const int l = t % L;

    float acc = cb[d];
#pragma unroll
    for (int k = 0; k < 4; ++k) {
        const int lp = l - 3 + k;
        if (lp >= 0)
            acc += xz[(size_t)(t - 3 + k) * (2 * di) + d] * cw[d * 4 + k];
    }
    const float v = silu_f(acc);
    xh[idx]  = v;
    xhh[idx] = (_Float16)v;
}

// ---------------------------------------------------------------------------
// delta = softplus(dt @ W_dt^T + b_dt); dt = first dr cols of DBC (row stride N2)
// K = dr <= 32 -> VALU
// ---------------------------------------------------------------------------
__global__ __launch_bounds__(256)
void delta_kernel(const float* __restrict__ dbc, const float* __restrict__ Wdt,
                  const float* __restrict__ bdt, float* __restrict__ delta,
                  int tokens, int di, int dr, int N2)
{
    const int idx = blockIdx.x * blockDim.x + threadIdx.x;
    if (idx >= tokens * di) return;
    const int d = idx % di;
    const int t = idx / di;
    const float* __restrict__ dt = dbc + (size_t)t * N2;
    float acc = bdt[d];
    for (int r = 0; r < dr; ++r)
        acc = fmaf(dt[r], Wdt[d * dr + r], acc);
    delta[idx] = softplus_f(acc);
}

// ---------------------------------------------------------------------------
// Selective scan fused with skip (xh*D) and SiLU(z) gate; f16 output (GEMM3 A).
// One lane per (b, d) channel; 16-state recurrence in VGPRs.
// ---------------------------------------------------------------------------
__global__ __launch_bounds__(128)
void scan_kernel(const float* __restrict__ delta, const float* __restrict__ dbc,
                 const float* __restrict__ xh, const float* __restrict__ xz,
                 const float* __restrict__ A_log, const float* __restrict__ Dp,
                 _Float16* __restrict__ yg, int L, int di, int dr, int N2)
{
    const int d = blockIdx.x * blockDim.x + threadIdx.x;
    const int b = blockIdx.y;
    if (d >= di) return;

    float A[16], h[16];
#pragma unroll
    for (int s = 0; s < 16; ++s) {
        A[s] = -__expf(A_log[d * 16 + s]);
        h[s] = 0.f;
    }
    const float Dd = Dp[d];

    for (int l = 0; l < L; ++l) {
        const size_t tok = (size_t)b * L + l;
        const float dl = delta[tok * di + d];
        const float xv = xh[tok * di + d];
        const float* __restrict__ bm = dbc + tok * N2 + dr;     // B_t[0..15]
        const float* __restrict__ cm = bm + 16;                 // C_t[0..15]
        const float dx = dl * xv;
        float y = 0.f;
#pragma unroll
        for (int s = 0; s < 16; ++s) {
            const float dA = __expf(dl * A[s]);
            h[s] = fmaf(dA, h[s], dx * bm[s]);
            y = fmaf(h[s], cm[s], y);
        }
        y = fmaf(xv, Dd, y);
        const float z = xz[tok * (size_t)(2 * di) + di + d];
        yg[tok * di + d] = (_Float16)(y * silu_f(z));
    }
}

// ---------------------------------------------------------------------------
// Stem conv 4x4 stride 4 (3 -> 64) + frozen-BN + ReLU. (4,3,224,224)->(4,64,56,56)
// ---------------------------------------------------------------------------
__global__ __launch_bounds__(256)
void stem_kernel(const float* __restrict__ x, const float* __restrict__ w,
                 const float* __restrict__ g, const float* __restrict__ b,
                 float* __restrict__ out)
{
    const int idx = blockIdx.x * blockDim.x + threadIdx.x;
    const int total = 4 * 64 * 56 * 56;
    if (idx >= total) return;
    const int wd = idx % 56;
    const int h  = (idx / 56) % 56;
    const int c  = (idx / (56 * 56)) % 64;
    const int bb = idx / (56 * 56 * 64);

    float acc = 0.f;
#pragma unroll
    for (int ic = 0; ic < 3; ++ic)
#pragma unroll
        for (int kh = 0; kh < 4; ++kh)
#pragma unroll
            for (int kw = 0; kw < 4; ++kw)
                acc = fmaf(x[((size_t)(bb * 3 + ic) * 224 + (4 * h + kh)) * 224 + (4 * wd + kw)],
                           w[((c * 3 + ic) * 4 + kh) * 4 + kw], acc);
    const float scale = g[c] * rsqrtf(1.f + LN_EPS);
    out[idx] = fmaxf(0.f, fmaf(acc, scale, b[c]));
}

// ---------------------------------------------------------------------------
// Downsample conv 2x2 stride 2 (Ci -> Co) + frozen-BN + ReLU.
// ---------------------------------------------------------------------------
__global__ __launch_bounds__(256)
void down_kernel(const float* __restrict__ x, const float* __restrict__ w,
                 const float* __restrict__ g, const float* __restrict__ b,
                 float* __restrict__ out, int Ci, int Co, int Hi, int Wi)
{
    const int Ho = Hi >> 1, Wo = Wi >> 1;
    const int idx = blockIdx.x * blockDim.x + threadIdx.x;
    const int total = 4 * Co * Ho * Wo;
    if (idx >= total) return;
    const int wd = idx % Wo;
    const int h  = (idx / Wo) % Ho;
    const int c  = (idx / (Wo * Ho)) % Co;
    const int bb = idx / (Wo * Ho * Co);

    float acc = 0.f;
    for (int ic = 0; ic < Ci; ++ic) {
        const float* __restrict__ xp = x + ((size_t)(bb * Ci + ic) * Hi + 2 * h) * Wi + 2 * wd;
        const float* __restrict__ wp = w + ((size_t)c * Ci + ic) * 4;
        acc = fmaf(xp[0],      wp[0], acc);
        acc = fmaf(xp[1],      wp[1], acc);
        acc = fmaf(xp[Wi],     wp[2], acc);
        acc = fmaf(xp[Wi + 1], wp[3], acc);
    }
    const float scale = g[c] * rsqrtf(1.f + LN_EPS);
    out[idx] = fmaxf(0.f, fmaf(acc, scale, b[c]));
}

// ---------------------------------------------------------------------------
// Host orchestration
// ---------------------------------------------------------------------------
extern "C" void kernel_launch(void* const* d_in, const int* in_sizes, int n_in,
                              void* d_out, int out_size, void* d_ws, size_t ws_size,
                              hipStream_t stream)
{
    (void)in_sizes; (void)n_in; (void)out_size; (void)ws_size;

    // d_in layout: [0]=x; params leaves in JAX sorted-key tree order:
    //  blocks (15 blocks x 11: A_log,D,W_dt,W_in,W_out,W_x,b_dt,conv_b,conv_w,ln_b,ln_g),
    //  down (3 x: b,g,w), stem_b, stem_g, stem_w
    const float* x_in = (const float*)d_in[0];
    auto P = [&](int i) { return (const float*)d_in[i]; };

    // Workspace carving (float units), worst-case across stages (~45 MB)
    float* ws    = (float*)d_ws;
    float* X0    = ws;                 // 802816   fp32  NCHW feature ping
    float* X1    = X0    + 802816;     // 802816   fp32  NCHW feature pong
    float* XZ    = X1    + 802816;     // 3211264  fp32  [M, 2di]
    float* XH    = XZ    + 3211264;    // 1605632  fp32  [M, di]   (scan input)
    float* DBC   = XH    + 1605632;    // 451584   fp32  [M, dr+32]
    float* DELTA = DBC   + 451584;     // 1605632  fp32  [M, di]
    _Float16* SLNh = (_Float16*)(DELTA + 1605632);   // 802816 h  [M, C]
    _Float16* XHh  = SLNh + 802816;                  // 1605632 h [M, di]
    _Float16* YGh  = XHh  + 1605632;                 // 1605632 h [M, di]
    _Float16* WH   = YGh  + 1605632;                 // 1638400 h (W_in|W_x|W_out f16)

    const int B = 4;
    const int CH[4]  = {64, 128, 256, 512};
    const int HH[4]  = {56, 28, 14, 7};
    const int DEP[4] = {2, 2, 9, 2};
    const int blk_start[4] = {0, 2, 4, 13};

    // stem: x -> X0
    {
        const float* sw = P(177);  // stem_w
        const float* sg = P(176);  // stem_g
        const float* sb = P(175);  // stem_b
        const int total = B * 64 * 56 * 56;
        stem_kernel<<<(total + 255) / 256, 256, 0, stream>>>(x_in, sw, sg, sb, X0);
    }

    float* xcur = X0;
    float* xalt = X1;
    size_t out_off = 0;

    for (int st = 0; st < 4; ++st) {
        const int C  = CH[st];
        const int H  = HH[st];
        const int L  = H * H;
        const int di = 2 * C;
        const int dr = C / 16;
        const int N2 = dr + 32;          // dr + 2*D_STATE
        const int M  = B * L;

        for (int bi = 0; bi < DEP[st]; ++bi) {
            const int base = 1 + 11 * (blk_start[st] + bi);
            const float* A_log  = P(base + 0);
            const float* Dp     = P(base + 1);
            const float* W_dt   = P(base + 2);
            const float* W_in   = P(base + 3);
            const float* W_out  = P(base + 4);
            const float* W_x    = P(base + 5);
            const float* b_dt   = P(base + 6);
            const float* conv_b = P(base + 7);
            const float* conv_w = P(base + 8);
            const float* ln_b   = P(base + 9);
            const float* ln_g   = P(base + 10);

            // 0. weights -> f16 workspace slices
            _Float16* Wih = WH;                       // [2di, C]
            _Float16* Wxh = Wih + (size_t)(2 * di) * C;   // [N2, di]
            _Float16* Woh = Wxh + (size_t)N2 * di;        // [C, di]
            {
                const int n1 = 2 * di * C, n2 = N2 * di, n3 = C * di;
                f32_to_f16_kernel<<<(n1 + 255) / 256, 256, 0, stream>>>(W_in,  Wih, n1);
                f32_to_f16_kernel<<<(n2 + 255) / 256, 256, 0, stream>>>(W_x,   Wxh, n2);
                f32_to_f16_kernel<<<(n3 + 255) / 256, 256, 0, stream>>>(W_out, Woh, n3);
            }

            // 1. LayerNorm (NCHW -> token-major f16)
            ln_kernel<<<(M * 32 + 255) / 256, 256, 0, stream>>>(xcur, ln_g, ln_b, SLNh, M, C, L);

            // 2. XZ = SLN @ W_in^T   [M, 2di]  (WMMA)
            launch_gemm(SLNh, Wih, XZ, nullptr, M, 2 * di, C, L, 0, 0, stream);

            // 3. causal depthwise conv + SiLU -> XH (fp32) / XHh (f16)
            {
                const int total = M * di;
                dwconv_silu_kernel<<<(total + 255) / 256, 256, 0, stream>>>(XZ, conv_w, conv_b,
                                                                            XH, XHh, B, L, di);
            }

            // 4. DBC = XH @ W_x^T    [M, dr+32]  (WMMA)
            launch_gemm(XHh, Wxh, DBC, nullptr, M, N2, di, L, 0, 0, stream);

            // 5. delta = softplus(dt @ W_dt^T + b_dt)  [M, di]
            {
                const int total = M * di;
                delta_kernel<<<(total + 255) / 256, 256, 0, stream>>>(DBC, W_dt, b_dt, DELTA,
                                                                      M, di, dr, N2);
            }

            // 6. selective scan + skip + gate -> YGh  [M, di] f16
            scan_kernel<<<dim3(di / 128, B), 128, 0, stream>>>(DELTA, DBC, XH, XZ, A_log, Dp, YGh,
                                                               L, di, dr, N2);

            // 7. residual: xcur += (YG @ W_out^T) scattered to NCHW  (WMMA, fused epilogue)
            launch_gemm(YGh, Woh, nullptr, xcur, M, C, di, L, C, 1, stream);
        }

        // emit stage features
        const size_t feat = (size_t)B * C * L;
        hipMemcpyAsync((float*)d_out + out_off, xcur, feat * sizeof(float),
                       hipMemcpyDeviceToDevice, stream);
        out_off += feat;

        // downsample into the other buffer
        if (st < 3) {
            const float* db = P(166 + 3 * st);
            const float* dg = P(167 + 3 * st);
            const float* dw = P(168 + 3 * st);
            const int Co = CH[st + 1];
            const int total = B * Co * (H / 2) * (H / 2);
            down_kernel<<<(total + 255) / 256, 256, 0, stream>>>(xcur, dw, dg, db, xalt, C, Co, H, H);
            float* t = xcur; xcur = xalt; xalt = t;
        }
    }
}
